// ASG_seq2seq_quantile_onehot_11012296147470
// MI455X (gfx1250) — matrix operand order repaired and checked by reading, not compiled, not run on hardware
//
#include <hip/hip_runtime.h>
#include <math.h>

#define N_    8192
#define F_    32
#define T_    24
#define H_    12
#define HSS_  768
#define G_    16
#define KNN_  8
#define NQ_   9
#define ZF_   95
#define DECIN_ 111
#define H4_   3072
#define KE_   800   /* 32 + 768 */
#define KD_   896   /* pad(111 + 768 = 879) -> 896 */
#define OUT1_ 384
#define CDIV(a,b) (((a)+(b)-1)/(b))

typedef unsigned short u16;
typedef __attribute__((ext_vector_type(16))) __bf16 v16bf;
typedef __attribute__((ext_vector_type(8)))  float  v8f;
typedef __attribute__((ext_vector_type(4))) unsigned int u32x4;
typedef __attribute__((ext_vector_type(8))) int i32x8;
typedef __attribute__((ext_vector_type(4))) int i32x4;

#ifdef __has_builtin
# if defined(__gfx1250__) && __has_builtin(__builtin_amdgcn_tensor_load_to_lds) && __has_builtin(__builtin_amdgcn_s_wait_tensorcnt)
#  define USE_TDM 1
# endif
#endif
#ifndef USE_TDM
# define USE_TDM 0
#endif

union FragAB { v16bf v; uint4 q[2]; };

__device__ __forceinline__ u16 f2bf(float x){
    unsigned u = __float_as_uint(x);
    u += 0x7FFFu + ((u >> 16) & 1u);
    return (u16)(u >> 16);
}
__device__ __forceinline__ float sigf(float x){ return 1.0f/(1.0f+__expf(-x)); }
__device__ __forceinline__ float lrelu(float x, float s){ return x > 0.0f ? x : x*s; }

#if USE_TDM
// Issue one TDM 2D tile load: tile 32 elems (bf16) x `rows` rows from gsrc
// (row stride ld elements) into LDS at lds_off, padding +4 DWORDs every 16
// DWORDs so rows land on an 80-byte LDS pitch ([rows][40] u16).
__device__ __forceinline__ void tdm_load_tile(const u16* gsrc, int ld,
                                              unsigned lds_off, unsigned rows,
                                              unsigned tdim0, unsigned tdim1){
    unsigned long long ga = (unsigned long long)(size_t)gsrc;
    u32x4 g0;
    g0.x = 1u;                                   // count=1 (valid descriptor)
    g0.y = lds_off;                              // LDS byte address
    g0.z = (unsigned)(ga & 0xFFFFFFFFu);         // global addr [31:0]
    g0.w = (unsigned)((ga >> 32) & 0x1FFFFFFu)   // global addr [56:32]
         | (2u << 30);                           // type=2 ("image")
    unsigned long long st0 = (unsigned long long)ld;  // dim0 stride (elements)
    i32x8 g1;
    g1[0] = (int)((1u<<16)      // data_size = 2 bytes
                | (1u<<20)      // pad_enable
                | (3u<<22)      // pad_interval: 16 DWORDs (64B) before padding
                | (3u<<25));    // pad_amount: 4 DWORDs (16B)
    g1[1] = (int)((tdim0 & 0xFFFFu) << 16);                      // tensor_dim0 lo
    g1[2] = (int)((tdim0 >> 16) | ((tdim1 & 0xFFFFu) << 16));    // dim0 hi | dim1 lo
    g1[3] = (int)((tdim1 >> 16) | (32u << 16));                  // dim1 hi | tile_dim0=32
    g1[4] = (int)(rows);                                         // tile_dim1=rows, tile_dim2=0
    g1[5] = (int)(st0 & 0xFFFFFFFFu);                            // dim0 stride lo
    g1[6] = (int)((st0 >> 32) & 0xFFFFu);                        // stride hi | dim1 stride lo=0
    g1[7] = 0;
    i32x4 z4; z4[0]=0; z4[1]=0; z4[2]=0; z4[3]=0;
    i32x8 z8; z8[0]=0; z8[1]=0; z8[2]=0; z8[3]=0; z8[4]=0; z8[5]=0; z8[6]=0; z8[7]=0;
    // 6-arg variant (clang-23 / therock headers): (g0, g1, g2, g3, g4, cpol)
    __builtin_amdgcn_tensor_load_to_lds(g0, g1, z4, z4, z8, 0);
}
#endif

// ---------------- init / weight prep ----------------
__global__ void k_zero_b(u16* p, long long n){
    long long i = (long long)blockIdx.x*blockDim.x + threadIdx.x;
    long long st = (long long)gridDim.x*blockDim.x;
    for (; i < n; i += st) p[i] = 0;
}
__global__ void k_zero_f(float* p, long long n){
    long long i = (long long)blockIdx.x*blockDim.x + threadIdx.x;
    long long st = (long long)gridDim.x*blockDim.x;
    for (; i < n; i += st) p[i] = 0.0f;
}
// dst[o*ld + k] = bf16( k<k1 ? w1[o,k] : k<k1+k2 ? w2[o,k-k1] : 0 )
__global__ void k_pack_w(const float* __restrict__ w1, int k1,
                         const float* __restrict__ w2, int k2,
                         u16* __restrict__ dst, int ld, int nout){
    long long tot = (long long)nout * ld;
    long long i = (long long)blockIdx.x*blockDim.x + threadIdx.x;
    long long st = (long long)gridDim.x*blockDim.x;
    for (; i < tot; i += st){
        int o = (int)(i / ld), k = (int)(i % ld);
        float v = 0.0f;
        if (k < k1)            v = w1[(long long)o*k1 + k];
        else if (k < k1 + k2)  v = w2[(long long)o*k2 + (k - k1)];
        dst[i] = f2bf(v);
    }
}
__global__ void k_addb(const float* a, const float* b, float* d, int n){
    int i = blockIdx.x*blockDim.x + threadIdx.x;
    if (i < n) d[i] = a[i] + b[i];
}

// ---------------- attribute embedding ----------------
__global__ void k_sl(const int* __restrict__ f,
                     const float* wa0, const float* wa1, const float* wa2, const float* wa3,
                     float* __restrict__ sl, float* __restrict__ outtail){
    int i = blockIdx.x*blockDim.x + threadIdx.x;
    if (i >= N_) return;
    float v0 = tanhf(wa0[f[i*4+0]]);
    float v1 = tanhf(wa1[f[i*4+1]]);
    float v2 = tanhf(wa2[f[i*4+2]]);
    float v3 = tanhf(wa3[f[i*4+3]]);
    sl[i*4+0]=v0; sl[i*4+1]=v1; sl[i*4+2]=v2; sl[i*4+3]=v3;
    outtail[i*4+0]=v0; outtail[i*4+1]=v1; outtail[i*4+2]=v2; outtail[i*4+3]=v3;
}

// ---------------- KNN (wave per row, per-lane top8 + shuffle merge) ----------------
__global__ __launch_bounds__(256) void k_knn(const float* __restrict__ sl,
                                             const int* __restrict__ batch,
                                             int* __restrict__ nbr, float* __restrict__ ew){
    int wid  = (blockIdx.x*blockDim.x + threadIdx.x) >> 5;
    int lane = threadIdx.x & 31;
    if (wid >= N_) return;
    const int i = wid;
    float a0 = sl[i*4+0], a1 = sl[i*4+1], a2 = sl[i*4+2], a3 = sl[i*4+3];
    int bi = batch[i];
    float bd[8]; int bj[8];
    #pragma unroll
    for (int k = 0; k < 8; ++k){ bd[k] = 3.0e38f; bj[k] = 0x7FFFFFFF; }
    for (int j = lane; j < N_; j += 32){
        if (batch[j] != bi) continue;
        float d0 = sl[j*4+0]-a0, d1 = sl[j*4+1]-a1, d2 = sl[j*4+2]-a2, d3 = sl[j*4+3]-a3;
        float d = d0*d0 + d1*d1 + d2*d2 + d3*d3;
        if (d < bd[7] || (d == bd[7] && j < bj[7])){
            bd[7] = d; bj[7] = j;
            #pragma unroll
            for (int k = 7; k >= 1; --k){
                if (bd[k] < bd[k-1] || (bd[k] == bd[k-1] && bj[k] < bj[k-1])){
                    float td = bd[k]; bd[k] = bd[k-1]; bd[k-1] = td;
                    int   tj = bj[k]; bj[k] = bj[k-1]; bj[k-1] = tj;
                }
            }
        }
    }
    int p = 0;
    for (int r = 0; r < 8; ++r){
        float d = (p < 8) ? bd[p] : 3.0e38f;
        int   j = (p < 8) ? bj[p] : 0x7FFFFFFF;
        #pragma unroll
        for (int off = 16; off > 0; off >>= 1){
            float dd = __shfl_xor(d, off);
            int   jj = __shfl_xor(j, off);
            if (dd < d || (dd == d && jj < j)){ d = dd; j = jj; }
        }
        if (p < 8 && bd[p] == d && bj[p] == j) ++p;
        if (lane == 0){
            nbr[i*8+r] = (j == 0x7FFFFFFF) ? i : j;
            ew[i*8+r]  = (j == 0x7FFFFFFF) ? 0.0f : __expf(-10.0f * d);
        }
    }
}

// ---------------- encoder input projection ----------------
__global__ void k_xe(const float* __restrict__ x, const float* __restrict__ Wx,
                     const float* __restrict__ bx, u16* __restrict__ xeb){
    int idx = blockIdx.x*blockDim.x + threadIdx.x;
    if (idx >= T_*N_) return;
    int t = idx / N_, n = idx % N_;
    float xv[F_];
    #pragma unroll
    for (int ff = 0; ff < F_; ++ff) xv[ff] = x[((long long)n*F_ + ff)*T_ + t];
    u16* d = xeb + (long long)idx * F_;
    #pragma unroll
    for (int g = 0; g < F_; ++g){
        float s = bx[g];
        #pragma unroll
        for (int ff = 0; ff < F_; ++ff) s += xv[ff]*Wx[g*F_+ff];
        d[g] = f2bf(tanhf(s));
    }
}
__global__ void k_pack_xe(const u16* __restrict__ xet, u16* __restrict__ encA){
    int n = blockIdx.x*blockDim.x + threadIdx.x;
    if (n >= N_) return;
    const uint4* s = (const uint4*)(xet + (long long)n*F_);
    uint4* d = (uint4*)(encA + (long long)n*KE_);
    d[0]=s[0]; d[1]=s[1]; d[2]=s[2]; d[3]=s[3];
}

// ---------------- bf16 WMMA GEMM: C[M,Nout] = A[M,K] * Bt[Nout,K]^T (+bias, act) -------
// Block tile 256x128, 8 waves each own a 32x128 strip (2x8 WMMA accumulators,
// so each B fragment feeds two WMMAs -> ~half the LDS reads per matmul).
// K double-buffered through LDS. On gfx1250 the next k-tiles are DMA'd by the
// Tensor Data Mover (wave 0 issues 2 tensor_load_to_lds, TENSORcnt-tracked)
// while all waves run WMMAs on the current tile.
__global__ __launch_bounds__(256) void k_gemm(
    const u16* __restrict__ A, int lda,
    const u16* __restrict__ Bt, int ldb,
    const float* __restrict__ bias,
    float* __restrict__ C, int ldc,
    int M, int Nout, int Kdim, int act, float slope)
{
    __shared__ __align__(16) u16 As[2][256][40];
    __shared__ __align__(16) u16 Bs[2][128][40];
    const int tid  = threadIdx.x;
    const int lane = tid & 31, wave = tid >> 5;
    const int hi = lane >> 4, l15 = lane & 15;
    const int mBlk = blockIdx.y * 256, nBlk = blockIdx.x * 128;
    const int nk = Kdim >> 5;

    v8f acc0[8], acc1[8];
    v8f zacc = {0.f,0.f,0.f,0.f,0.f,0.f,0.f,0.f};
    #pragma unroll
    for (int i = 0; i < 8; ++i){ acc0[i] = zacc; acc1[i] = zacc; }

#if USE_TDM
    const u16* aTile = A  + (size_t)mBlk*lda;
    const u16* bTile = Bt + (size_t)nBlk*ldb;
    unsigned ldsA[2], ldsB[2];
    ldsA[0] = (unsigned)(size_t)&As[0][0][0];
    ldsA[1] = (unsigned)(size_t)&As[1][0][0];
    ldsB[0] = (unsigned)(size_t)&Bs[0][0][0];
    ldsB[1] = (unsigned)(size_t)&Bs[1][0][0];
    if (wave == 0){
        tdm_load_tile(aTile, lda, ldsA[0], 256u, (unsigned)Kdim, (unsigned)M);
        tdm_load_tile(bTile, ldb, ldsB[0], 128u, (unsigned)Kdim, (unsigned)Nout);
        __builtin_amdgcn_s_wait_tensorcnt(0);
    }
#else
    const int rb = tid >> 1, hsel = tid & 1;
    { // tile 0: A 256 rows (one per thread), B 128 rows (half-row per thread)
        const uint4* ga = (const uint4*)(A + (size_t)(mBlk + tid)*lda);
        uint4* da = (uint4*)&As[0][tid][0];
        da[0]=ga[0]; da[1]=ga[1]; da[2]=ga[2]; da[3]=ga[3];
        const uint4* gb = (const uint4*)(Bt + (size_t)(nBlk + rb)*ldb + hsel*16);
        uint4* db = (uint4*)&Bs[0][rb][hsel*16];
        db[0]=gb[0]; db[1]=gb[1];
    }
#endif
    __syncthreads();
    for (int ks = 0; ks < nk; ++ks){
        const int cur = ks & 1;
        const int nxt = cur ^ 1;
        const int kb = (ks+1) << 5;
#if USE_TDM
        if (ks + 1 < nk && wave == 0){
            tdm_load_tile(aTile + kb, lda, ldsA[nxt], 256u, (unsigned)Kdim, (unsigned)M);
            tdm_load_tile(bTile + kb, ldb, ldsB[nxt], 128u, (unsigned)Kdim, (unsigned)Nout);
        }
#else
        if (ks + 1 < nk){
            const uint4* ga = (const uint4*)(A + (size_t)(mBlk + tid)*lda + kb);
            uint4* da = (uint4*)&As[nxt][tid][0];
            da[0]=ga[0]; da[1]=ga[1]; da[2]=ga[2]; da[3]=ga[3];
            const uint4* gb = (const uint4*)(Bt + (size_t)(nBlk + rb)*ldb + kb + hsel*16);
            uint4* db = (uint4*)&Bs[nxt][rb][hsel*16];
            db[0]=gb[0]; db[1]=gb[1];
            if (ks + 2 < nk){
                __builtin_prefetch((const void*)(A  + (size_t)(mBlk + tid)*lda + (kb+32)), 0, 0);
                __builtin_prefetch((const void*)(Bt + (size_t)(nBlk + rb)*ldb + (kb+32) + hsel*16), 0, 0);
            }
        }
#endif
        // Two A fragments (rows wave*32 .. wave*32+31); ISA 16-bit A layout
        FragAB af0, af1;
        const int mrow0 = wave*32 + l15;
        const int mrow1 = mrow0 + 16;
        af0.q[0] = *(const uint4*)&As[cur][mrow0][8*hi];        // k = 8hi..8hi+7
        af0.q[1] = *(const uint4*)&As[cur][mrow0][16 + 8*hi];   // k = 16+8hi..23+8hi
        af1.q[0] = *(const uint4*)&As[cur][mrow1][8*hi];
        af1.q[1] = *(const uint4*)&As[cur][mrow1][16 + 8*hi];
        // Each B fragment feeds two WMMAs (one per A sub-tile).
        #pragma unroll
        for (int nt = 0; nt < 8; ++nt){
            FragAB bfr;
            const int ncol = nt*16 + l15;
            bfr.q[0] = *(const uint4*)&Bs[cur][ncol][8*hi];       // k = 8hi..8hi+7
            bfr.q[1] = *(const uint4*)&Bs[cur][ncol][16 + 8*hi];  // k = 16+8hi..23+8hi
            acc0[nt] = __builtin_amdgcn_wmma_f32_16x16x32_bf16(
                false, af0.v, false, bfr.v, (short)0, acc0[nt], false, false);
            acc1[nt] = __builtin_amdgcn_wmma_f32_16x16x32_bf16(
                false, af1.v, false, bfr.v, (short)0, acc1[nt], false, false);
        }
#if USE_TDM
        if (ks + 1 < nk && wave == 0) __builtin_amdgcn_s_wait_tensorcnt(0);
#endif
        __syncthreads();
    }
    // epilogue: C/D layout -> VGPR rr: rows rr (lanes 0-15) / rr+8 (lanes 16-31), col = l15
    #pragma unroll
    for (int nt = 0; nt < 8; ++nt){
        const int nc = nBlk + nt*16 + l15;
        const float bv = bias ? bias[nc] : 0.0f;
        #pragma unroll
        for (int rr = 0; rr < 8; ++rr){
            const int mr0 = mBlk + wave*32 + rr + hi*8;
            float v0 = acc0[nt][rr] + bv;
            if (act) v0 = lrelu(v0, slope);
            C[(size_t)mr0*ldc + nc] = v0;
            const int mr1 = mr0 + 16;
            float v1 = acc1[nt][rr] + bv;
            if (act) v1 = lrelu(v1, slope);
            C[(size_t)mr1*ldc + nc] = v1;
        }
    }
}

// ---------------- LSTM elementwise update ----------------
__global__ void k_lstm(const float* __restrict__ gates, float* __restrict__ c,
                       u16* __restrict__ adst, int ald, u16* __restrict__ hb){
    int idx = blockIdx.x*blockDim.x + threadIdx.x;
    if (idx >= N_*HSS_) return;
    int n = idx / HSS_, j = idx % HSS_;
    const float* g = gates + (size_t)n*H4_;
    float gi = g[j], gf = g[HSS_+j], gg = g[2*HSS_+j], go = g[3*HSS_+j];
    float cv = c[idx];
    float c2 = sigf(gf)*cv + sigf(gi)*tanhf(gg);
    float h2 = sigf(go)*tanhf(c2);
    c[idx] = c2;
    u16 hb16 = f2bf(h2);
    adst[(size_t)n*ald + j] = hb16;
    if (hb) hb[idx] = hb16;
}
__global__ void k_packlin0(const float* __restrict__ hs, u16* __restrict__ decA){
    int idx = blockIdx.x*blockDim.x + threadIdx.x;
    if (idx >= N_*HSS_) return;
    int n = idx / HSS_, j = idx % HSS_;
    decA[(size_t)n*KD_ + DECIN_ + j] = f2bf(hs[idx]);
}

// ---------------- graph weights softmax + hg init ----------------
__global__ void k_wsoft(const int* __restrict__ mask, const int* __restrict__ nbr,
                        const float* __restrict__ ew, const float* __restrict__ z,
                        float* __restrict__ w, float* __restrict__ hg, int t){
    int i = blockIdx.x*blockDim.x + threadIdx.x;
    if (i >= N_) return;
    int mi = mask[i*H_+t];
    float e[8]; int val[8]; float mx = -1e30f; int any = 0;
    #pragma unroll
    for (int k = 0; k < 8; ++k){
        int j = nbr[i*8+k];
        int v = (mi != 0) && (mask[j*H_+t] != 0);
        val[k] = v; e[k] = ew[i*8+k];
        if (v){ any = 1; mx = fmaxf(mx, e[k]); }
    }
    float s = 0.0f, p[8];
    #pragma unroll
    for (int k = 0; k < 8; ++k){ p[k] = val[k] ? __expf(e[k]-mx) : 0.0f; s += p[k]; }
    float inv = (any && s > 0.0f) ? 1.0f/s : 0.0f;
    #pragma unroll
    for (int k = 0; k < 8; ++k) w[i*8+k] = p[k]*inv;
    #pragma unroll
    for (int g = 0; g < G_; ++g)
        hg[i*G_+g] = (g < 4) ? z[(size_t)i*(F_-1)*H_ + g*H_ + t] : 0.0f;
}

// ---------------- GatedGraphConv ----------------
__global__ void k_ggcm(const float* __restrict__ hg, const float* __restrict__ ggcw,
                       float* __restrict__ m, int l){
    int idx = blockIdx.x*blockDim.x + threadIdx.x;
    if (idx >= N_*G_) return;
    int i = idx >> 4, g = idx & 15;
    const float* wl = ggcw + l*G_*G_;
    float s = 0.0f;
    #pragma unroll
    for (int q = 0; q < G_; ++q) s += hg[i*G_+q]*wl[q*G_+g];
    m[idx] = s;
}
__global__ void k_ggcgru(const float* __restrict__ m, const int* __restrict__ nbr,
                         const float* __restrict__ w,
                         const float* __restrict__ wih, const float* __restrict__ whh,
                         const float* __restrict__ bih, const float* __restrict__ bhh,
                         float* __restrict__ hg, u16* __restrict__ decA){
    int i = blockIdx.x*blockDim.x + threadIdx.x;
    if (i >= N_) return;
    float agg[G_];
    #pragma unroll
    for (int g = 0; g < G_; ++g) agg[g] = 0.0f;
    for (int k = 0; k < 8; ++k){
        float wk = w[i*8+k];
        const float* mj = m + (size_t)nbr[i*8+k]*G_;
        #pragma unroll
        for (int g = 0; g < G_; ++g) agg[g] += wk*mj[g];
    }
    float hv[G_];
    #pragma unroll
    for (int g = 0; g < G_; ++g) hv[g] = hg[i*G_+g];
    #pragma unroll
    for (int g = 0; g < G_; ++g){
        float gr = bih[g], gz = bih[G_+g], gn = bih[2*G_+g];
        float hr = bhh[g], hz = bhh[G_+g], hn = bhh[2*G_+g];
        #pragma unroll
        for (int q = 0; q < G_; ++q){
            gr += agg[q]*wih[g*G_+q];
            gz += agg[q]*wih[(G_+g)*G_+q];
            gn += agg[q]*wih[(2*G_+g)*G_+q];
            hr += hv[q]*whh[g*G_+q];
            hz += hv[q]*whh[(G_+g)*G_+q];
            hn += hv[q]*whh[(2*G_+g)*G_+q];
        }
        float rr = sigf(gr+hr), zz = sigf(gz+hz);
        float nv = tanhf(gn + rr*hn);
        float h2 = (1.0f-zz)*nv + zz*hv[g];
        hg[i*G_+g] = h2;
        if (decA) decA[(size_t)i*KD_ + ZF_ + g] = f2bf(lrelu(h2, 0.01f));
    }
}

// ---------------- xz = tanh(zcat_t @ Wz^T + bz) -> bf16 into decA cols [0,95) ----------
__global__ void k_xz(const float* __restrict__ hs, const float* __restrict__ z,
                     const float* __restrict__ Wz, const float* __restrict__ bz,
                     u16* __restrict__ decA, int t){
    long long idx = (long long)blockIdx.x*blockDim.x + threadIdx.x;
    if (idx >= (long long)N_*ZF_) return;
    int i = (int)(idx / ZF_), j = (int)(idx % ZF_);
    float s = bz[j];
    const float* wr = Wz + (size_t)j*ZF_;
    #pragma unroll 4
    for (int cidx = 0; cidx < 64; ++cidx) s += hs[(size_t)i*HSS_ + cidx*H_ + t]*wr[cidx];
    #pragma unroll 4
    for (int cidx = 64; cidx < ZF_; ++cidx) s += z[(size_t)i*(F_-1)*H_ + (cidx-64)*H_ + t]*wr[cidx];
    decA[(size_t)i*KD_ + j] = f2bf(tanhf(s));
}

// ---------------- final small head: y2 = leaky(y1 @ w2^T + b2, 0.1) ----------------
__global__ void k_out2(const float* __restrict__ y1, const float* __restrict__ w2,
                       const float* __restrict__ b2, float* __restrict__ out, int t){
    int idx = blockIdx.x*blockDim.x + threadIdx.x;
    if (idx >= N_*NQ_) return;
    int i = idx / NQ_, q = idx % NQ_;
    float s = b2[q];
    const float* wr = w2 + q*OUT1_;
    const float* yr = y1 + (size_t)i*OUT1_;
    #pragma unroll 8
    for (int cidx = 0; cidx < OUT1_; ++cidx) s += yr[cidx]*wr[cidx];
    out[(size_t)i*NQ_*H_ + q*H_ + t] = lrelu(s, 0.1f);
}

extern "C" void kernel_launch(void* const* d_in, const int* in_sizes, int n_in,
                              void* d_out, int out_size, void* d_ws, size_t ws_size,
                              hipStream_t stream) {
    const float* x       = (const float*)d_in[0];
    const int*   f       = (const int*)  d_in[1];
    const int*   batch   = (const int*)  d_in[2];
    const float* z       = (const float*)d_in[3];
    const int*   mask    = (const int*)  d_in[4];
    const float* wa0     = (const float*)d_in[5];
    const float* wa1     = (const float*)d_in[6];
    const float* wa2     = (const float*)d_in[7];
    const float* wa3     = (const float*)d_in[8];
    const float* Wx      = (const float*)d_in[9];
    const float* bx      = (const float*)d_in[10];
    const float* Wz      = (const float*)d_in[11];
    const float* bz      = (const float*)d_in[12];
    const float* ggcw    = (const float*)d_in[13];
    const float* gru_wih = (const float*)d_in[14];
    const float* gru_whh = (const float*)d_in[15];
    const float* gru_bih = (const float*)d_in[16];
    const float* gru_bhh = (const float*)d_in[17];
    const float* enc_wih = (const float*)d_in[18];
    const float* enc_whh = (const float*)d_in[19];
    const float* enc_bih = (const float*)d_in[20];
    const float* enc_bhh = (const float*)d_in[21];
    const float* dec_wih = (const float*)d_in[22];
    const float* dec_whh = (const float*)d_in[23];
    const float* dec_bih = (const float*)d_in[24];
    const float* dec_bhh = (const float*)d_in[25];
    const float* lin0_w  = (const float*)d_in[26];
    const float* lin0_b  = (const float*)d_in[27];
    const float* out_w1  = (const float*)d_in[28];
    const float* out_b1  = (const float*)d_in[29];
    const float* out_w2  = (const float*)d_in[30];
    const float* out_b2  = (const float*)d_in[31];
    float* outp = (float*)d_out;

    char* wsb = (char*)d_ws;
    size_t off = 0;
    auto alloc = [&](size_t bytes)->char* {
        char* p = wsb + off;
        off += (bytes + 255) & ~(size_t)255;
        return p;
    };
    float* sl    = (float*)alloc((size_t)N_*4*4);
    int*   nbr   = (int*)  alloc((size_t)N_*KNN_*4);
    float* ew    = (float*)alloc((size_t)N_*KNN_*4);
    u16*   Wenc  = (u16*)  alloc((size_t)H4_*KE_*2);
    u16*   Wdec  = (u16*)  alloc((size_t)H4_*KD_*2);
    u16*   Wl0   = (u16*)  alloc((size_t)HSS_*HSS_*2);
    u16*   Wo1   = (u16*)  alloc((size_t)OUT1_*HSS_*2);
    float* biasE = (float*)alloc((size_t)H4_*4);
    float* biasD = (float*)alloc((size_t)H4_*4);
    u16*   xeb   = (u16*)  alloc((size_t)T_*N_*F_*2);
    u16*   encA  = (u16*)  alloc((size_t)N_*KE_*2);
    u16*   decA  = (u16*)  alloc((size_t)N_*KD_*2);
    float* gates = (float*)alloc((size_t)N_*H4_*4);
    float* cbuf  = (float*)alloc((size_t)N_*HSS_*4);
    float* hsnap = (float*)alloc((size_t)N_*HSS_*4);
    u16*   hb    = (u16*)  alloc((size_t)N_*HSS_*2);
    float* y1    = (float*)alloc((size_t)N_*OUT1_*4);
    float* hg    = (float*)alloc((size_t)N_*G_*4);
    float* mbuf  = (float*)alloc((size_t)N_*G_*4);
    float* wbuf  = (float*)alloc((size_t)N_*KNN_*4);
    (void)ws_size; (void)in_sizes; (void)n_in; (void)out_size;

    const int B = 256;
    // init
    k_zero_b<<<1024, B, 0, stream>>>(encA, (long long)N_*KE_);
    k_zero_b<<<1024, B, 0, stream>>>(decA, (long long)N_*KD_);
    k_zero_f<<<1024, B, 0, stream>>>(cbuf, (long long)N_*HSS_);
    // weight prep (bf16, [Nout,K] = B^T layout)
    k_pack_w<<<2048, B, 0, stream>>>(enc_wih, F_,    enc_whh, HSS_, Wenc, KE_, H4_);
    k_pack_w<<<2048, B, 0, stream>>>(dec_wih, DECIN_, dec_whh, HSS_, Wdec, KD_, H4_);
    k_pack_w<<<1024, B, 0, stream>>>(lin0_w, HSS_, (const float*)nullptr, 0, Wl0, HSS_, HSS_);
    k_pack_w<<< 512, B, 0, stream>>>(out_w1, HSS_, (const float*)nullptr, 0, Wo1, HSS_, OUT1_);
    k_addb<<<CDIV(H4_,B), B, 0, stream>>>(enc_bih, enc_bhh, biasE, H4_);
    k_addb<<<CDIV(H4_,B), B, 0, stream>>>(dec_bih, dec_bhh, biasD, H4_);
    // embeddings + knn + xe
    k_sl <<<CDIV(N_,B), B, 0, stream>>>(f, wa0, wa1, wa2, wa3, sl, outp + (size_t)N_*NQ_*H_);
    k_knn<<<N_/8, B, 0, stream>>>(sl, batch, nbr, ew);
    k_xe <<<CDIV(T_*N_,B), B, 0, stream>>>(x, Wx, bx, xeb);

    dim3 gGate(H4_/128, N_/256);   // (24, 32)
    dim3 gLin0(HSS_/128, N_/256);  // (6, 32)
    dim3 gOut1(OUT1_/128, N_/256); // (3, 32)

    // ---------------- encoder: 24 LSTM steps ----------------
    for (int t = 0; t < T_; ++t){
        k_pack_xe<<<CDIV(N_,B), B, 0, stream>>>(xeb + (size_t)t*N_*F_, encA);
        k_gemm<<<gGate, B, 0, stream>>>(encA, KE_, Wenc, KE_, biasE, gates, H4_,
                                        N_, H4_, KE_, 0, 0.0f);
        k_lstm<<<CDIV(N_*HSS_,B), B, 0, stream>>>(gates, cbuf, encA + F_, KE_, (u16*)nullptr);
    }
    // lin0 + leaky, snapshot h for zcat, seed decoder A-buffer
    k_gemm<<<gLin0, B, 0, stream>>>(encA + F_, KE_, Wl0, HSS_, lin0_b, hsnap, HSS_,
                                    N_, HSS_, HSS_, 1, 0.01f);
    k_packlin0<<<CDIV(N_*HSS_,B), B, 0, stream>>>(hsnap, decA);

    // ---------------- decoder: 12 steps ----------------
    for (int t = 0; t < H_; ++t){
        k_wsoft<<<CDIV(N_,B), B, 0, stream>>>(mask, nbr, ew, z, wbuf, hg, t);
        for (int l = 0; l < 3; ++l){
            k_ggcm  <<<CDIV(N_*G_,B), B, 0, stream>>>(hg, ggcw, mbuf, l);
            k_ggcgru<<<CDIV(N_,B),    B, 0, stream>>>(mbuf, nbr, wbuf, gru_wih, gru_whh,
                                                      gru_bih, gru_bhh, hg,
                                                      (l == 2) ? decA : (u16*)nullptr);
        }
        k_xz<<<CDIV(N_*ZF_,B), B, 0, stream>>>(hsnap, z, Wz, bz, decA, t);
        k_gemm<<<gGate, B, 0, stream>>>(decA, KD_, Wdec, KD_, biasD, gates, H4_,
                                        N_, H4_, KD_, 0, 0.0f);
        k_lstm<<<CDIV(N_*HSS_,B), B, 0, stream>>>(gates, cbuf, decA + DECIN_, KD_, hb);
        k_gemm<<<gOut1, B, 0, stream>>>(hb, HSS_, Wo1, HSS_, out_b1, y1, OUT1_,
                                        N_, OUT1_, HSS_, 1, 0.01f);
        k_out2<<<CDIV(N_*NQ_,B), B, 0, stream>>>(y1, out_w2, out_b2, outp, t);
    }
}
